// Neighborhood3DAttentionLayer_TStyle_37460704756337
// MI455X (gfx1250) — matrix operand order repaired and checked by reading, not compiled, not run on hardware
//
#include <hip/hip_runtime.h>
#include <hip/hip_bf16.h>
#include <math.h>

// ---- problem constants (from reference) ----
#define DIMC   768
#define NHEADS 16
#define EMBD   1024
#define DH     48
#define TT     3
#define RESN   24
#define HWSZ   (RESN*RESN)      // 576
#define THW    (TT*HWSZ)        // 1728
#define KTN    3
#define KHN    5
#define KWN    5
#define NBR    (KTN*KHN*KWN)    // 75
#define MLPD   (4*DIMC)         // 3072
#define NFUSED (3*DIMC+MLPD)    // 5376
#define KCAT   (DIMC+MLPD)      // 3840
#define EPSF   1e-5f

typedef __attribute__((ext_vector_type(16))) __bf16 v16bf;
typedef __attribute__((ext_vector_type(8)))  __bf16 v8bf;
typedef __attribute__((ext_vector_type(8)))  float  v8f;

// ---- CDNA5 async global->LDS copy (ASYNCcnt-tracked) ----
__device__ __forceinline__ void async_b128_to_lds(unsigned lds_off, const void* gptr) {
    unsigned long long ga = (unsigned long long)(uintptr_t)gptr;
    asm volatile("global_load_async_to_lds_b128 %0, %1, off"
                 :: "v"(lds_off), "v"(ga) : "memory");
}
__device__ __forceinline__ void wait_async0() {
    asm volatile("s_wait_asynccnt 0x0" ::: "memory");
}

// ---------------- RoPE cos/sin tables (1728 x 24) ----------------
__global__ void rope_tab_kernel(float* __restrict__ cosb, float* __restrict__ sinb) {
    int p = blockIdx.x;
    int j = threadIdx.x;
    if (j >= DH / 2) return;
    int t = p / HWSZ, hw = p % HWSZ, h = hw / RESN, w = hw % RESN;
    int axis = j / 8, jj = j % 8;
    float coord = (axis == 0) ? (float)t : (axis == 1) ? (float)h : (float)w;
    float inv = powf(10000.f, -(float)jj / 8.f);   // 10000^(-2*jj/16)
    float a = coord * inv;
    cosb[p * (DH / 2) + j] = cosf(a);
    sinb[p * (DH / 2) + j] = sinf(a);
}

// ---------------- mod = silu(emb) @ w_mod + b_mod  (BT x 1536) ----------------
__global__ __launch_bounds__(256)
void mod_kernel(const float* __restrict__ emb, const float* __restrict__ w_mod,
                const float* __restrict__ b_mod, float* __restrict__ mod) {
    __shared__ float se[EMBD];
    int bt = blockIdx.x;
    for (int e = threadIdx.x; e < EMBD; e += 256) {
        float v = emb[(size_t)bt * EMBD + e];
        se[e] = v / (1.f + expf(-v));
    }
    __syncthreads();
    for (int j = threadIdx.x; j < 2 * DIMC; j += 256) {
        float acc = b_mod[j];
        for (int e = 0; e < EMBD; e++) acc += se[e] * w_mod[(size_t)e * (2 * DIMC) + j];
        mod[(size_t)bt * (2 * DIMC) + j] = acc;
    }
}

// ---------------- h = LN(xf)*(1+scale)+shift -> bf16 (Mtot x 768) ----------------
__global__ __launch_bounds__(256)
void ln_mod_kernel(const float* __restrict__ x, const float* __restrict__ mod,
                   __bf16* __restrict__ hbf) {
    int p  = blockIdx.x;
    int b  = p / THW, wi = p % THW;
    int t  = wi / HWSZ, hw = wi % HWSZ;
    const float* xr = x + ((size_t)(b * TT + t) * DIMC) * HWSZ + hw;  // xf[p,c] = xr[c*HWSZ]
    float xs[3];
    float s = 0.f, s2 = 0.f;
#pragma unroll
    for (int i = 0; i < 3; i++) {
        int c = threadIdx.x + i * 256;
        float v = xr[(size_t)c * HWSZ];
        xs[i] = v; s += v; s2 += v * v;
    }
    __shared__ float red[256], red2[256];
    red[threadIdx.x] = s; red2[threadIdx.x] = s2;
    __syncthreads();
    for (int st = 128; st > 0; st >>= 1) {
        if (threadIdx.x < st) {
            red[threadIdx.x]  += red[threadIdx.x + st];
            red2[threadIdx.x] += red2[threadIdx.x + st];
        }
        __syncthreads();
    }
    float mean = red[0] / (float)DIMC;
    float var  = red2[0] / (float)DIMC - mean * mean;
    float inv  = rsqrtf(var + EPSF);
    const float* sc = mod + (size_t)(b * TT + t) * (2 * DIMC);
#pragma unroll
    for (int i = 0; i < 3; i++) {
        int c = threadIdx.x + i * 256;
        float yv = (xs[i] - mean) * inv * (1.f + sc[c]) + sc[DIMC + c];
        hbf[(size_t)p * DIMC + c] = (__bf16)yv;
    }
}

// ---------------- fp32 -> bf16 bulk convert ----------------
__global__ void cvt_bf16_kernel(const float* __restrict__ src, __bf16* __restrict__ dst, size_t n) {
    size_t i = (size_t)blockIdx.x * blockDim.x + threadIdx.x;
    if (i < n) dst[i] = (__bf16)src[i];
}

// ---------------- WMMA bf16 GEMM ----------------
// Block: 256 threads / 8 waves laid out 4(M) x 2(N): block tile 64M x 128N.
// B tile (32K x 128N, 8 KB) is staged to LDS with global_load_async_to_lds_b128
// (double-buffered, XOR-swizzled 16B chunks for bank-conflict-free ds reads,
// shared by all 8 waves).  A fragments are register-pipelined global b128 loads.
// Per K-step a wave loads all 4 B fragments first, then issues 4 WMMAs so the
// compiler can use partial s_wait_dscnt and overlap LDS latency with compute.
// mode 0: C[row*ldc+col] = acc + bias[col]
// mode 1: out[(bt*DIMC+col)*HWSZ+hw] = skip_x[idx] + acc + bias[col] (transposed)
__global__ __launch_bounds__(256)
void gemm_bf16_kernel(const __bf16* __restrict__ A, const __bf16* __restrict__ Bm,
                      int K, int lda, int ldb,
                      const float* __restrict__ bias,
                      float* __restrict__ C, int ldc,
                      const float* __restrict__ skip_x, float* __restrict__ out,
                      int mode) {
    __shared__ __align__(16) unsigned char btile[2][32 * 256];   // 2 x 8 KB stages

    const int lane   = threadIdx.x & 31;
    const int wave   = threadIdx.x >> 5;
    const int wave_n = wave & 1;        // 2 waves across N
    const int wave_m = wave >> 1;       // 4 waves across M
    const int tnb    = blockIdx.x * 128;             // block N base
    const int tn     = tnb + wave_n * 64;            // wave N base
    const int tm     = blockIdx.y * 64 + wave_m * 16;
    const int arow   = lane & 15;
    const int kb0    = (lane >> 4) * 8;  // A-frag K sub-base per ISA 16-bit A layout

    const unsigned ldsbase = (unsigned)(uintptr_t)&btile[0][0];

    // cooperatively stage one 32x128 bf16 B tile (8 KB) as 512 async b128 chunks,
    // 16B chunk-in-row XOR-swizzled by (row & 15) to spread LDS banks.
    auto issueB = [&](int kb, int stage) {
#pragma unroll
        for (int r = 0; r < 2; ++r) {
            int c   = threadIdx.x + r * 256;  // chunk id 0..511
            int row = c >> 4;                 // 16 x 16B chunks per 256B row
            int cio = c & 15;
            const __bf16* g = Bm + (size_t)(kb + row) * ldb + tnb + cio * 8;
            unsigned loff = ldsbase + stage * 8192u + row * 256u
                          + ((unsigned)(cio ^ (row & 15)) * 16u);
            async_b128_to_lds(loff, g);
        }
    };

    v8f acc[4];
#pragma unroll
    for (int i = 0; i < 4; i++) acc[i] = (v8f){0.f,0.f,0.f,0.f,0.f,0.f,0.f,0.f};

    const int steps = K / 32;

    // prologue: stage 0 B tile + first A fragment
    issueB(0, 0);
    v8bf alo = *(const v8bf*)(A + (size_t)(tm + arow) * lda + kb0);
    v8bf ahi = *(const v8bf*)(A + (size_t)(tm + arow) * lda + kb0 + 16);
    wait_async0();
    __syncthreads();

    const int sw = lane & 15;           // unused; reads swizzle by (k & 15), row == k

    for (int s = 0; s < steps; ++s) {
        const int cur = s & 1;
        const int nxt = cur ^ 1;
        v8bf nlo, nhi;
        if (s + 1 < steps) {
            int kb = (s + 1) * 32;
            issueB(kb, nxt);                               // async, overlaps compute
            nlo = *(const v8bf*)(A + (size_t)(tm + arow) * lda + kb + kb0);
            nhi = *(const v8bf*)(A + (size_t)(tm + arow) * lda + kb + kb0 + 16);
        }

        v16bf afrag;
#pragma unroll
        for (int i = 0; i < 8; i++) { afrag[i] = alo[i]; afrag[8 + i] = ahi[i]; }

        // load ALL B fragments first (64 bf16 ds reads per lane), then WMMA x4
        v16bf bfr[4];
#pragma unroll
        for (int i = 0; i < 4; i++) {
            int n = wave_n * 64 + 16 * i + (lane & 15);
#pragma unroll
            for (int e = 0; e < 16; e++) {
                int k = (e < 8) ? (kb0 + e) : (16 + kb0 + (e - 8));
                const unsigned char* rowp = &btile[cur][k * 256];
                bfr[i][e] = *(const __bf16*)(rowp + (((n >> 3) ^ (k & 15)) * 16) + (n & 7) * 2);
            }
        }
#pragma unroll
        for (int i = 0; i < 4; i++) {
            acc[i] = __builtin_amdgcn_wmma_f32_16x16x32_bf16(
                false, afrag, false, bfr[i], (short)0, acc[i], false, false);
        }

        if (s + 1 < steps) {
            wait_async0();          // next B stage resident
            __syncthreads();        // visible to all waves; cur stage reads done
            alo = nlo; ahi = nhi;
        }
    }

    const int r0 = (lane >> 4) * 8;   // C/D layout: lane<16 -> M=j, lane>=16 -> M=8+j
    const int cn = lane & 15;
#pragma unroll
    for (int i = 0; i < 4; i++) {
        int col = tn + 16 * i + cn;
        float bv = bias[col];
#pragma unroll
        for (int j = 0; j < 8; j++) {
            int grow = tm + r0 + j;
            float v = acc[i][j] + bv;
            if (mode == 0) {
                C[(size_t)grow * ldc + col] = v;
            } else {
                int b = grow / THW, wi = grow % THW;
                int t = wi / HWSZ, hw = wi % HWSZ;
                size_t idx = ((size_t)(b * TT + t) * DIMC + col) * HWSZ + hw;
                out[idx] = skip_x[idx] + v;
            }
        }
    }
}

// ---------------- q/k LN + RoPE + scale, v copy; layout [head][p][dh] ----------------
__global__ __launch_bounds__(256)
void qkv_kernel(const float* __restrict__ fused,
                const float* __restrict__ qn_w, const float* __restrict__ qn_b,
                const float* __restrict__ kn_w, const float* __restrict__ kn_b,
                const float* __restrict__ cosb, const float* __restrict__ sinb,
                float* __restrict__ qb, float* __restrict__ kb, float* __restrict__ vb,
                int Mtot) {
    int gid = blockIdx.x * blockDim.x + threadIdx.x;
    if (gid >= Mtot * NHEADS * 3) return;
    int s    = gid % 3;
    int head = (gid / 3) % NHEADS;
    int p    = gid / (3 * NHEADS);
    const float* src = fused + (size_t)p * NFUSED + s * DIMC + head * DH;
    float* dst = ((s == 0) ? qb : (s == 1) ? kb : vb) + ((size_t)head * Mtot + p) * DH;
    if (s == 2) {
#pragma unroll
        for (int d = 0; d < DH; d++) dst[d] = src[d];
        return;
    }
    float y[DH];
    float sum = 0.f, sum2 = 0.f;
#pragma unroll
    for (int d = 0; d < DH; d++) { float v = src[d]; y[d] = v; sum += v; sum2 += v * v; }
    float mean = sum / (float)DH;
    float var  = sum2 / (float)DH - mean * mean;
    float inv  = rsqrtf(var + EPSF);
    const float* wn = (s == 0) ? qn_w : kn_w;
    const float* bn = (s == 0) ? qn_b : kn_b;
#pragma unroll
    for (int d = 0; d < DH; d++) y[d] = (y[d] - mean) * inv * wn[d] + bn[d];
    int ppos = p % THW;
    const float* cs = cosb + (size_t)ppos * (DH / 2);
    const float* sn = sinb + (size_t)ppos * (DH / 2);
    float scale = (s == 0) ? rsqrtf((float)DH) : 1.f;
#pragma unroll
    for (int i = 0; i < DH / 2; i++) {
        float x1 = y[2 * i], x2 = y[2 * i + 1];
        float c = cs[i], ss = sn[i];
        dst[2 * i]     = (x1 * c - x2 * ss) * scale;
        dst[2 * i + 1] = (x1 * ss + x2 * c) * scale;
    }
}

// ---------------- silu(mlp_h) -> bf16 into acat[:, 768:3840] ----------------
__global__ void mlp_silu_kernel(const float* __restrict__ fused, __bf16* __restrict__ acat,
                                int Mtot) {
    size_t i = (size_t)blockIdx.x * blockDim.x + threadIdx.x;
    size_t n = (size_t)Mtot * MLPD;
    if (i >= n) return;
    size_t p = i / MLPD, j = i % MLPD;
    float v = fused[p * NFUSED + 3 * DIMC + j];
    acat[p * KCAT + DIMC + j] = (__bf16)(v / (1.f + expf(-v)));
}

// ---------------- neighborhood attention: wave per (token, head) ----------------
__global__ __launch_bounds__(128)
void attn_kernel(const float* __restrict__ qb, const float* __restrict__ kbuf,
                 const float* __restrict__ vb, __bf16* __restrict__ acat, int Mtot) {
    __shared__ float lg[4][NBR + 1];
    int lane = threadIdx.x & 31;
    int wave = threadIdx.x >> 5;
    int p    = blockIdx.x;
    int head = blockIdx.y * 4 + wave;
    int b = p / THW, wi = p % THW;
    int t = wi / HWSZ, hw = wi % HWSZ, h = hw / RESN, w = hw % RESN;
    int ts  = min(max(t - KTN / 2, 0), TT  - KTN);
    int hs  = min(max(h - KHN / 2, 0), RESN - KHN);
    int ws2 = min(max(w - KWN / 2, 0), RESN - KWN);
    const float* qp = qb + ((size_t)head * Mtot + p) * DH;
    int d0 = lane, d1 = lane + 32;
    bool has2 = lane < (DH - 32);           // lanes 0..15 also carry d+32
    float qa  = qp[d0];
    float qc  = has2 ? qp[d1] : 0.f;
    for (int n = 0; n < NBR; n++) {
        int a = n / (KHN * KWN), rc = n % (KHN * KWN), c = rc / KWN, e = rc % KWN;
        int pp = b * THW + (ts + a) * HWSZ + (hs + c) * RESN + (ws2 + e);
        const float* kp = kbuf + ((size_t)head * Mtot + pp) * DH;
        float part = qa * kp[d0] + (has2 ? qc * kp[d1] : 0.f);
#pragma unroll
        for (int off = 16; off >= 1; off >>= 1) part += __shfl_xor(part, off, 32);
        if (lane == 0) lg[wave][n] = part;
    }
    __syncthreads();
    float mx = -1e30f;
#pragma unroll
    for (int r = 0; r < 3; r++) { int n = lane + 32 * r; if (n < NBR) mx = fmaxf(mx, lg[wave][n]); }
#pragma unroll
    for (int off = 16; off >= 1; off >>= 1) mx = fmaxf(mx, __shfl_xor(mx, off, 32));
    float ev[3]; float sum = 0.f;
#pragma unroll
    for (int r = 0; r < 3; r++) {
        int n = lane + 32 * r;
        ev[r] = (n < NBR) ? expf(lg[wave][n] - mx) : 0.f;
        sum += ev[r];
    }
#pragma unroll
    for (int off = 16; off >= 1; off >>= 1) sum += __shfl_xor(sum, off, 32);
    __syncthreads();
#pragma unroll
    for (int r = 0; r < 3; r++) { int n = lane + 32 * r; if (n < NBR) lg[wave][n] = ev[r]; }
    __syncthreads();
    float invs = 1.f / sum;
    float acca = 0.f, accb = 0.f;
    for (int n = 0; n < NBR; n++) {
        int a = n / (KHN * KWN), rc = n % (KHN * KWN), c = rc / KWN, e = rc % KWN;
        int pp = b * THW + (ts + a) * HWSZ + (hs + c) * RESN + (ws2 + e);
        const float* vp = vb + ((size_t)head * Mtot + pp) * DH;
        float wgt = lg[wave][n] * invs;
        acca += wgt * vp[d0];
        if (has2) accb += wgt * vp[d1];
    }
    __bf16* dst = acat + (size_t)p * KCAT + head * DH;   // xa transposed -> channel head*48+d
    dst[d0] = (__bf16)acca;
    if (has2) dst[d1] = (__bf16)accb;
}

// ---------------- host orchestration ----------------
extern "C" void kernel_launch(void* const* d_in, const int* in_sizes, int n_in,
                              void* d_out, int out_size, void* d_ws, size_t ws_size,
                              hipStream_t stream) {
    const float* x       = (const float*)d_in[0];
    const float* emb     = (const float*)d_in[1];
    const float* w_mod   = (const float*)d_in[2];
    const float* b_mod   = (const float*)d_in[3];
    const float* qn_w    = (const float*)d_in[4];
    const float* qn_b    = (const float*)d_in[5];
    const float* kn_w    = (const float*)d_in[6];
    const float* kn_b    = (const float*)d_in[7];
    const float* W_fused = (const float*)d_in[8];
    const float* b_fused = (const float*)d_in[9];
    const float* W_out   = (const float*)d_in[10];
    const float* W_mlp   = (const float*)d_in[11];
    const float* b_mlp   = (const float*)d_in[12];
    float* out = (float*)d_out;

    int BT   = in_sizes[1] / EMBD;   // B*T
    int Bn   = BT / TT;
    int Mtot = Bn * THW;

    char* ws = (char*)d_ws;
    size_t off = 0;
    auto alloc = [&](size_t bytes) -> char* {
        char* p = ws + off;
        off = (off + bytes + 255) & ~(size_t)255;
        return p;
    };
    float*  cosb  = (float*) alloc((size_t)THW * (DH / 2) * sizeof(float));
    float*  sinb  = (float*) alloc((size_t)THW * (DH / 2) * sizeof(float));
    float*  modb  = (float*) alloc((size_t)BT * 2 * DIMC * sizeof(float));
    __bf16* hbf   = (__bf16*)alloc((size_t)Mtot * DIMC * sizeof(__bf16));
    __bf16* wfbf  = (__bf16*)alloc((size_t)DIMC * NFUSED * sizeof(__bf16));
    float*  fused = (float*) alloc((size_t)Mtot * NFUSED * sizeof(float));
    float*  qbuf  = (float*) alloc((size_t)NHEADS * Mtot * DH * sizeof(float));
    float*  kbuf  = (float*) alloc((size_t)NHEADS * Mtot * DH * sizeof(float));
    float*  vbuf  = (float*) alloc((size_t)NHEADS * Mtot * DH * sizeof(float));
    __bf16* acat  = (__bf16*)alloc((size_t)Mtot * KCAT * sizeof(__bf16));
    __bf16* wcat  = (__bf16*)alloc((size_t)KCAT * DIMC * sizeof(__bf16));
    (void)ws_size; (void)n_in; (void)out_size;

    // 1) RoPE tables
    rope_tab_kernel<<<dim3(THW), 32, 0, stream>>>(cosb, sinb);
    // 2) modulation
    mod_kernel<<<dim3(BT), 256, 0, stream>>>(emb, w_mod, b_mod, modb);
    // 3) weight conversions (wcat = [W_out ; W_mlp] row-concat over K)
    {
        size_t n1 = (size_t)DIMC * NFUSED;
        size_t n2 = (size_t)DIMC * DIMC;
        size_t n3 = (size_t)MLPD * DIMC;
        cvt_bf16_kernel<<<(unsigned)((n1 + 255) / 256), 256, 0, stream>>>(W_fused, wfbf, n1);
        cvt_bf16_kernel<<<(unsigned)((n2 + 255) / 256), 256, 0, stream>>>(W_out, wcat, n2);
        cvt_bf16_kernel<<<(unsigned)((n3 + 255) / 256), 256, 0, stream>>>(W_mlp, wcat + n2, n3);
    }
    // 4) LN + modulation -> bf16 activations
    ln_mod_kernel<<<dim3(Mtot), 256, 0, stream>>>(x, modb, hbf);
    // 5) big fused GEMM: (Mtot x 768) @ (768 x 5376) + b_fused  [WMMA bf16 + async LDS]
    gemm_bf16_kernel<<<dim3(NFUSED / 128, Mtot / 64), 256, 0, stream>>>(
        hbf, wfbf, DIMC, DIMC, NFUSED, b_fused, fused, NFUSED,
        nullptr, nullptr, 0);
    // 6) q/k LN + RoPE + scale, v copy
    {
        int tot = Mtot * NHEADS * 3;
        qkv_kernel<<<(tot + 255) / 256, 256, 0, stream>>>(
            fused, qn_w, qn_b, kn_w, kn_b, cosb, sinb, qbuf, kbuf, vbuf, Mtot);
    }
    // 7) silu(mlp_h) -> acat[:, 768:]
    {
        size_t n = (size_t)Mtot * MLPD;
        mlp_silu_kernel<<<(unsigned)((n + 255) / 256), 256, 0, stream>>>(fused, acat, Mtot);
    }
    // 8) neighborhood attention -> acat[:, :768] (bf16)
    attn_kernel<<<dim3(Mtot, NHEADS / 4), 128, 0, stream>>>(qbuf, kbuf, vbuf, acat, Mtot);
    // 9) combined output GEMM: (Mtot x 3840) @ (3840 x 768) + b_mlp + skip, transposed store
    gemm_bf16_kernel<<<dim3(DIMC / 128, Mtot / 64), 256, 0, stream>>>(
        acat, wcat, KCAT, KCAT, DIMC, b_mlp, nullptr, 0,
        x, out, 1);
}
